// _MultiheadAttention_11682311045245
// MI455X (gfx1250) — matrix operand order repaired and checked by reading, not compiled, not run on hardware
//
#include <hip/hip_runtime.h>
#include <hip/hip_bf16.h>

// ---------------------------------------------------------------------------
// Multi-head attention forward for MI455X (gfx1250, wave32, WMMA bf16).
// D = A*B + C via v_wmma_f32_16x16x32_bf16; fp32 accumulation throughout.
// ---------------------------------------------------------------------------

typedef __attribute__((ext_vector_type(16))) __bf16 v16bf;
typedef __attribute__((ext_vector_type(8)))  __bf16 v8bf;
typedef __attribute__((ext_vector_type(8)))  float  v8f;

static constexpr int BB = 2;      // batch
static constexpr int SS = 2048;   // sequence length
static constexpr int DM = 1024;   // d_model
static constexpr int NH = 16;     // heads
static constexpr int DK = 64;     // head dim (d_k == d_v)
static constexpr int BS = BB * SS;        // 4096 rows
static constexpr float SCALE = 0.125f;    // 1/sqrt(64)

#define DEV __device__ __forceinline__

union BF16x16 { v16bf v; v8bf h[2]; __bf16 e[16]; };

// A-fragment (16x32 bf16, row-major source, lda elements):
//   lanes 0-15  : row M = l, K = {k0..k0+7, k0+16..k0+23}
//   lanes 16-31 : row M = l-16, K = {k0+8..k0+15, k0+24..k0+31}
DEV v16bf load_a_bf16(const __bf16* __restrict__ A, int lda, int row0, int k0, int lane) {
  const int l16 = lane & 15;
  const int hi  = (lane >> 4) & 1;
  const __bf16* p = A + (size_t)(row0 + l16) * lda + k0 + hi * 8;
  BF16x16 u;
  u.h[0] = *(const v8bf*)p;
  u.h[1] = *(const v8bf*)(p + 16);
  return u.v;
}

// Same A layout but fp32 source, converted to bf16 in-register.
DEV v16bf load_a_f32(const float* __restrict__ A, int lda, int row0, int k0, int lane) {
  const int l16 = lane & 15;
  const int hi  = (lane >> 4) & 1;
  const float* p = A + (size_t)(row0 + l16) * lda + k0 + hi * 8;
  v8f lo = *(const v8f*)p;
  v8f hv = *(const v8f*)(p + 16);
  BF16x16 u;
#pragma unroll
  for (int i = 0; i < 8; ++i) { u.e[i] = (__bf16)lo[i]; u.e[8 + i] = (__bf16)hv[i]; }
  return u.v;
}

// B-fragment (32x16 bf16). Source is stored "B-rows": row n holds B[:,n]
// contiguously over K (ldb elements). lanes 0-15: K={k0..k0+7, k0+16..k0+23},
// lanes 16-31: K={k0+8..k0+15, k0+24..k0+31}  -> two 16-byte loads per lane.
DEV v16bf load_b_rows(const __bf16* __restrict__ Br, int ldb, int n0, int k0, int lane) {
  const int l16 = lane & 15;
  const int hi  = (lane >> 4) & 1;
  const __bf16* p = Br + (size_t)(n0 + l16) * ldb + k0 + hi * 8;
  BF16x16 u;
  u.h[0] = *(const v8bf*)p;
  u.h[1] = *(const v8bf*)(p + 16);
  return u.v;
}

DEV v8f wmma_bf16(v16bf a, v16bf b, v8f c) {
  return __builtin_amdgcn_wmma_f32_16x16x32_bf16(
      /*neg_a=*/false, a, /*neg_b=*/false, b,
      /*c_mod=*/(short)0, c, /*reuse_a=*/false, /*reuse_b=*/false);
}

// ---------------------------------------------------------------------------
// fp32 -> bf16 elementwise conversion (grid-stride)
// ---------------------------------------------------------------------------
__global__ void cvt_f32_to_bf16_kernel(const float* __restrict__ src,
                                       __bf16* __restrict__ dst, int n) {
  int i = blockIdx.x * blockDim.x + threadIdx.x;
  const int stride = gridDim.x * blockDim.x;
  for (; i < n; i += stride) dst[i] = (__bf16)src[i];
}

// ---------------------------------------------------------------------------
// Generic bf16 GEMM: C[M,N] = A[M,K] * Brows[N,K]^T + bias
// Block = 128 threads = 4 waves (2x2), each wave computes a 64x64 tile
// (4x4 grid of 16x16 WMMA accumulators). M,N multiples of 128; K of 32.
// STORE_MODE 0: fp32 row-major [M,N]
// STORE_MODE 1: bf16 head-split [b,h,s,d]   (rows = b*SS+s, cols = h*DK+d)
// STORE_MODE 2: bf16 head-split transposed [b,h,d,s]
// ---------------------------------------------------------------------------
template <int STORE_MODE>
__global__ void __launch_bounds__(128)
gemm_bf16_kernel(const __bf16* __restrict__ A, const __bf16* __restrict__ Brows,
                 const float* __restrict__ bias, void* __restrict__ Cout,
                 int M, int N, int K) {
  const int lane = threadIdx.x & 31;
  const int wave = threadIdx.x >> 5;
  const int tileM = blockIdx.y * 128 + (wave >> 1) * 64;
  const int tileN = blockIdx.x * 128 + (wave & 1) * 64;

  v8f acc[4][4] = {};

  for (int k0 = 0; k0 < K; k0 += 32) {
    v16bf af[4], bfm[4];
#pragma unroll
    for (int i = 0; i < 4; ++i) af[i] = load_a_bf16(A, K, tileM + i * 16, k0, lane);
#pragma unroll
    for (int j = 0; j < 4; ++j) bfm[j] = load_b_rows(Brows, K, tileN + j * 16, k0, lane);
    if (k0 + 32 < K) {
      __builtin_prefetch(A + (size_t)(tileM + (lane & 15)) * K + k0 + 32, 0, 1);
      __builtin_prefetch(Brows + (size_t)(tileN + (lane & 15)) * K + k0 + 32, 0, 1);
    }
#pragma unroll
    for (int i = 0; i < 4; ++i)
#pragma unroll
      for (int j = 0; j < 4; ++j) acc[i][j] = wmma_bf16(af[i], bfm[j], acc[i][j]);
  }

  const int l16 = lane & 15;
  const int hi  = (lane >> 4) & 1;
#pragma unroll
  for (int i = 0; i < 4; ++i) {
#pragma unroll
    for (int j = 0; j < 4; ++j) {
      const int n = tileN + j * 16 + l16;
      const float bv = bias ? bias[n] : 0.0f;
#pragma unroll
      for (int r = 0; r < 8; ++r) {
        const int m = tileM + i * 16 + r + hi * 8;
        const float val = acc[i][j][r] + bv;
        if (STORE_MODE == 0) {
          ((float*)Cout)[(size_t)m * N + n] = val;
        } else {
          const int b = m / SS, s = m % SS;
          const int h = n / DK, d = n % DK;
          __bf16* O = (__bf16*)Cout;
          if (STORE_MODE == 1)
            O[(((size_t)(b * NH + h)) * SS + s) * DK + d] = (__bf16)val;
          else
            O[(((size_t)(b * NH + h)) * DK + d) * SS + s] = (__bf16)val;
        }
      }
    }
  }
}

// ---------------------------------------------------------------------------
// Scores: attn[bh, s, t] = scale * (q[bh] . k[bh]^T) + prev[bh]  (fp32 out)
// per (b,h): M=N=SS, K=DK. grid=(SS/128, SS/128, B*H), block=128.
// ---------------------------------------------------------------------------
__global__ void __launch_bounds__(128)
scores_gemm_kernel(const __bf16* __restrict__ qh, const __bf16* __restrict__ kh,
                   const float* __restrict__ prev, float* __restrict__ attn) {
  const int lane = threadIdx.x & 31;
  const int wave = threadIdx.x >> 5;
  const int bh = blockIdx.z;
  const int tileM = blockIdx.y * 128 + (wave >> 1) * 64;
  const int tileN = blockIdx.x * 128 + (wave & 1) * 64;

  const __bf16* Aq = qh + (size_t)bh * SS * DK;
  const __bf16* Bk = kh + (size_t)bh * SS * DK;
  const float*  Pv = prev + (size_t)bh * SS * SS;
  float*        Ov = attn + (size_t)bh * SS * SS;

  v8f acc[4][4] = {};
#pragma unroll
  for (int k0 = 0; k0 < DK; k0 += 32) {
    v16bf af[4], bfm[4];
#pragma unroll
    for (int i = 0; i < 4; ++i) af[i] = load_a_bf16(Aq, DK, tileM + i * 16, k0, lane);
#pragma unroll
    for (int j = 0; j < 4; ++j) bfm[j] = load_b_rows(Bk, DK, tileN + j * 16, k0, lane);
#pragma unroll
    for (int i = 0; i < 4; ++i)
#pragma unroll
      for (int j = 0; j < 4; ++j) acc[i][j] = wmma_bf16(af[i], bfm[j], acc[i][j]);
  }

  const int l16 = lane & 15;
  const int hi  = (lane >> 4) & 1;
#pragma unroll
  for (int i = 0; i < 4; ++i)
#pragma unroll
    for (int j = 0; j < 4; ++j) {
      const int n = tileN + j * 16 + l16;
#pragma unroll
      for (int r = 0; r < 8; ++r) {
        const int m = tileM + i * 16 + r + hi * 8;
        const size_t idx = (size_t)m * SS + n;
        Ov[idx] = acc[i][j][r] * SCALE + Pv[idx];
      }
    }
}

// ---------------------------------------------------------------------------
// Row-wise softmax in place over attn (rows of SS=2048 fp32).
// One 256-thread block per row; 8 elements per thread held in registers.
// ---------------------------------------------------------------------------
__global__ void __launch_bounds__(256)
softmax_rows_kernel(float* __restrict__ attn) {
  const int tid = threadIdx.x;
  float* rp = attn + (size_t)blockIdx.x * SS;
  const int base = tid * 8;

  float4 a = *(const float4*)(rp + base);
  float4 b = *(const float4*)(rp + base + 4);
  float x[8] = {a.x, a.y, a.z, a.w, b.x, b.y, b.z, b.w};

  __shared__ float red[256];
  float m = x[0];
#pragma unroll
  for (int i = 1; i < 8; ++i) m = fmaxf(m, x[i]);
  red[tid] = m;
  __syncthreads();
  for (int s = 128; s > 0; s >>= 1) {
    if (tid < s) red[tid] = fmaxf(red[tid], red[tid + s]);
    __syncthreads();
  }
  m = red[0];
  __syncthreads();

  float sum = 0.0f;
#pragma unroll
  for (int i = 0; i < 8; ++i) { x[i] = __expf(x[i] - m); sum += x[i]; }
  red[tid] = sum;
  __syncthreads();
  for (int s = 128; s > 0; s >>= 1) {
    if (tid < s) red[tid] += red[tid + s];
    __syncthreads();
  }
  const float inv = 1.0f / red[0];

  a = make_float4(x[0] * inv, x[1] * inv, x[2] * inv, x[3] * inv);
  b = make_float4(x[4] * inv, x[5] * inv, x[6] * inv, x[7] * inv);
  *(float4*)(rp + base) = a;
  *(float4*)(rp + base + 4) = b;
}

// ---------------------------------------------------------------------------
// P*V: Hc[b, s, h*DK+dv] = sum_t attn[bh,s,t] * v[bh,t,dv]
// attn read fp32 from d_out and converted to bf16 fragments in-register;
// v supplied transposed ([b,h,dv,t]) so B-fragments are contiguous over K.
// per (b,h): M=SS, N=DK, K=SS. grid=(SS/128, B*H), block=64 (2 waves).
// ---------------------------------------------------------------------------
__global__ void __launch_bounds__(64)
av_gemm_kernel(const float* __restrict__ attn, const __bf16* __restrict__ vT,
               __bf16* __restrict__ Hc) {
  const int lane = threadIdx.x & 31;
  const int wave = threadIdx.x >> 5;  // 0..1
  const int bh = blockIdx.y;
  const int b = bh / NH, h = bh % NH;
  const int tileM = blockIdx.x * 128 + wave * 64;

  const float*  P  = attn + (size_t)bh * SS * SS;
  const __bf16* Vt = vT + (size_t)bh * DK * SS;

  v8f acc[4][4] = {};
  for (int k0 = 0; k0 < SS; k0 += 32) {
    v16bf af[4], bfm[4];
#pragma unroll
    for (int i = 0; i < 4; ++i) af[i] = load_a_f32(P, SS, tileM + i * 16, k0, lane);
#pragma unroll
    for (int j = 0; j < 4; ++j) bfm[j] = load_b_rows(Vt, SS, j * 16, k0, lane);
    if (k0 + 32 < SS)
      __builtin_prefetch(P + (size_t)(tileM + (lane & 15)) * SS + k0 + 32, 0, 1);
#pragma unroll
    for (int i = 0; i < 4; ++i)
#pragma unroll
      for (int j = 0; j < 4; ++j) acc[i][j] = wmma_bf16(af[i], bfm[j], acc[i][j]);
  }

  const int l16 = lane & 15;
  const int hi  = (lane >> 4) & 1;
#pragma unroll
  for (int i = 0; i < 4; ++i)
#pragma unroll
    for (int j = 0; j < 4; ++j) {
      const int dv = j * 16 + l16;
#pragma unroll
      for (int r = 0; r < 8; ++r) {
        const int s = tileM + i * 16 + r + hi * 8;
        Hc[((size_t)b * SS + s) * DM + h * DK + dv] = (__bf16)acc[i][j][r];
      }
    }
}

// ---------------------------------------------------------------------------
extern "C" void kernel_launch(void* const* d_in, const int* in_sizes, int n_in,
                              void* d_out, int out_size, void* d_ws, size_t ws_size,
                              hipStream_t stream) {
  (void)in_sizes; (void)n_in; (void)out_size; (void)ws_size;

  const float* Q      = (const float*)d_in[0];
  const float* K      = (const float*)d_in[1];
  const float* V      = (const float*)d_in[2];
  const float* prev   = (const float*)d_in[3];
  const float* WQw    = (const float*)d_in[4];
  const float* WQbias = (const float*)d_in[5];
  const float* WKw    = (const float*)d_in[6];
  const float* WKbias = (const float*)d_in[7];
  const float* WVw    = (const float*)d_in[8];
  const float* WVbias = (const float*)d_in[9];
  const float* WOw    = (const float*)d_in[10];
  const float* WObias = (const float*)d_in[11];

  float* out  = (float*)d_out;                 // [B,S,DM] = 4,194,304 floats
  float* attn = out + (size_t)BS * DM;         // [B,H,S,S]

  // workspace carve-out (~67 MB)
  char* ws = (char*)d_ws;
  size_t off = 0;
  auto alloc_bf = [&](size_t elems) -> __bf16* {
    __bf16* p = (__bf16*)(ws + off);
    off += ((elems * sizeof(__bf16)) + 255) & ~(size_t)255;
    return p;
  };
  __bf16* Qb   = alloc_bf((size_t)BS * DM);
  __bf16* Kb   = alloc_bf((size_t)BS * DM);
  __bf16* Vb   = alloc_bf((size_t)BS * DM);
  __bf16* WQb  = alloc_bf((size_t)DM * DM);
  __bf16* WKb  = alloc_bf((size_t)DM * DM);
  __bf16* WVb  = alloc_bf((size_t)DM * DM);
  __bf16* WOb  = alloc_bf((size_t)DM * DM);
  __bf16* qh   = alloc_bf((size_t)BB * NH * SS * DK);  // [b,h,s,d]
  __bf16* kh   = alloc_bf((size_t)BB * NH * SS * DK);  // [b,h,s,d]
  __bf16* vT   = alloc_bf((size_t)BB * NH * DK * SS);  // [b,h,d,s]
  __bf16* Hc   = alloc_bf((size_t)BS * DM);            // [b,s, h*DK+dv]

  auto cvt = [&](const float* s, __bf16* d, int n) {
    cvt_f32_to_bf16_kernel<<<dim3(1024), dim3(256), 0, stream>>>(s, d, n);
  };
  cvt(Q, Qb, BS * DM);
  cvt(K, Kb, BS * DM);
  cvt(V, Vb, BS * DM);
  cvt(WQw, WQb, DM * DM);
  cvt(WKw, WKb, DM * DM);
  cvt(WVw, WVb, DM * DM);
  cvt(WOw, WOb, DM * DM);

  // Projections: [4096,1024] x [1024,1024]^T
  dim3 gProj(DM / 128, BS / 128);  // (8, 32)
  gemm_bf16_kernel<1><<<gProj, 128, 0, stream>>>(Qb, WQb, WQbias, qh, BS, DM, DM);
  gemm_bf16_kernel<1><<<gProj, 128, 0, stream>>>(Kb, WKb, WKbias, kh, BS, DM, DM);
  gemm_bf16_kernel<2><<<gProj, 128, 0, stream>>>(Vb, WVb, WVbias, vT, BS, DM, DM);

  // Scores + prev bias -> attn section of d_out (pre-softmax)
  dim3 gScore(SS / 128, SS / 128, BB * NH);  // (16,16,32)
  scores_gemm_kernel<<<gScore, 128, 0, stream>>>(qh, kh, prev, attn);

  // In-place row softmax (final attn_weights output)
  softmax_rows_kernel<<<dim3(BB * NH * SS), 256, 0, stream>>>(attn);

  // P*V -> concat heads (bf16)
  av_gemm_kernel<<<dim3(SS / 128, BB * NH), 64, 0, stream>>>(attn, vT, Hc);

  // Output projection -> fp32 out section of d_out
  gemm_bf16_kernel<0><<<gProj, 128, 0, stream>>>(Hc, WOb, WObias, out, BS, DM, DM);
}
